// GraphFlow_Polar_CNN_22471268892733
// MI455X (gfx1250) — matrix operand order repaired and checked by reading, not compiled, not run on hardware
//
#include <hip/hip_runtime.h>
#include <math.h>

typedef __attribute__((ext_vector_type(16))) _Float16 v16h;
typedef __attribute__((ext_vector_type(8)))  _Float16 v8h;
typedef __attribute__((ext_vector_type(2)))  _Float16 v2h;
typedef __attribute__((ext_vector_type(8)))  float    v8f;

#define HWP 1000   // H*W density plane (50x20)
#define C1P 736    // conv1 out: 46*16
#define C2P 504    // conv2 out: 42*12
#define C3P 304    // conv3 out: 38*8

// ---------------- CSR build ----------------

__global__ void k_init_cnt(int* cnt, int n) {
  int i = blockIdx.x * blockDim.x + threadIdx.x;
  if (i < n) cnt[i] = 1;  // self loop contributes one edge per node
}

__global__ void k_count(const long long* __restrict__ edges, int* cnt, int e) {
  int i = blockIdx.x * blockDim.x + threadIdx.x;
  if (i < e) atomicAdd(&cnt[(int)edges[(size_t)e + i]], 1);  // row 1 = targets
}

__global__ __launch_bounds__(1024) void k_scan(const int* __restrict__ cnt,
                                               int* __restrict__ off,
                                               int* __restrict__ cur, int n) {
  __shared__ int sh[1024];
  __shared__ int carry_s;
  int tid = threadIdx.x;
  if (tid == 0) carry_s = 0;
  __syncthreads();
  for (int base = 0; base < n; base += 1024) {
    int v = (base + tid < n) ? cnt[base + tid] : 0;
    sh[tid] = v;
    __syncthreads();
    for (int d = 1; d < 1024; d <<= 1) {
      int add = (tid >= d) ? sh[tid - d] : 0;
      __syncthreads();
      sh[tid] += add;
      __syncthreads();
    }
    int incl  = sh[tid];
    int carry = carry_s;
    if (base + tid < n) {
      int ex = carry + incl - v;
      off[base + tid] = ex;
      cur[base + tid] = ex;
    }
    __syncthreads();
    if (tid == 1023) carry_s = carry + incl;
    __syncthreads();
  }
  if (tid == 0) off[n] = carry_s;
}

__global__ void k_fill(const long long* __restrict__ edges, int* cur,
                       int* __restrict__ elist, int e, int n) {
  int i = blockIdx.x * blockDim.x + threadIdx.x;
  if (i < e + n) {
    int s, tg;
    if (i < e) { s = (int)edges[i]; tg = (int)edges[(size_t)e + i]; }
    else       { s = i - e; tg = s; }
    int p = atomicAdd(&cur[tg], 1);
    elist[p] = s;
  }
}

// ---------------- helpers ----------------

// 16 halves from two 16B-aligned LDS runs of 8 -> one WMMA operand
__device__ __forceinline__ v16h ld_frag(const _Float16* lo8, const _Float16* hi8) {
  v8h lo = *(const v8h*)lo8;
  v8h hi = *(const v8h*)hi8;
  return __builtin_shufflevector(lo, hi, 0, 1, 2, 3, 4, 5, 6, 7,
                                         8, 9, 10, 11, 12, 13, 14, 15);
}

// conv1 A-fragment: two rows of 4 channel-pairs (8 x b32, merged to 2addr)
__device__ __forceinline__ v16h ld_a1frag(const v2h* xsp2, int xb0) {
  v16h af;
#pragma unroll
  for (int i = 0; i < 4; ++i) {
    v2h a = xsp2[xb0 + i];
    v2h b = xsp2[xb0 + 24 + i];
    af[2 * i]     = a[0]; af[2 * i + 1] = a[1];
    af[8 + 2 * i] = b[0]; af[9 + 2 * i] = b[1];
  }
  return af;
}

// branch-free tanh: 1 - 2/(exp2(2x*log2e)+1); native exp/rcp rate, saturates
// correctly at +-1; plenty of accuracy ahead of f16 storage.
__device__ __forceinline__ float fast_tanh(float x) {
  float e = __builtin_amdgcn_exp2f(x * 2.885390043258667f);
  return 1.f - 2.f * __builtin_amdgcn_rcpf(e + 1.f);
}

// ---------------- fused per-node pipeline ----------------
//
// conv1: time channel folded into bias -> K = 2ch * 25 taps, taps padded to
//        8/kernel-row (48 tap slots, 96 K) -> 3 ksteps of wmma 16x16x32.
//        Input staged position-major, channel-pair packed: xsp[56][24][2].
// conv2: tap-pair decomposition: 13 pairs, K=32 (2 taps x 16ch). conv1 output
//        stored position-major a1p[752][16] so A-fragments are 2x ds_load_b128.

__global__ __launch_bounds__(256) void k_fused(
    const float* __restrict__ tptr, const float* __restrict__ data,
    const float* __restrict__ posd,
    const float* __restrict__ W1, const float* __restrict__ b1,
    const float* __restrict__ W2, const float* __restrict__ b2,
    const float* __restrict__ W3, const float* __restrict__ b3,
    const float* __restrict__ Wl, const float* __restrict__ bl,
    const int* __restrict__ off, const int* __restrict__ elist,
    float* __restrict__ out) {
  __shared__ __align__(16) _Float16 xsp[56 * 24 * 2];   //  5.4 KB padded input grid
  __shared__ __align__(16) _Float16 a1p[752 * 16];      // 24.1 KB conv1 act, pos-major
  __shared__ __align__(16) _Float16 w1p[16 * 96];       //  3.0 KB conv1 weights packed
  __shared__ __align__(16) _Float16 wb2[13 * 16 * 32];  // 13.3 KB conv2 tap-pair weights
  __shared__ float a2[5 * C2P];                         // 10.1 KB conv2 activations
  __shared__ float a3[C3P];                             //  1.2 KB conv3 activations
  __shared__ float bias1_s[16];
  __shared__ float w3s[128];

  const int tid  = threadIdx.x;
  const int node = blockIdx.x;
  const float tval = tptr[0];

  // ---- phase 0: zero pads + pack weights (disjoint LDS writes, no sync yet)
  {
    int* xw = (int*)xsp;                       // zero whole padded grid
    for (int i = tid; i < 56 * 24; i += 256) xw[i] = 0;
    int* aw = (int*)a1p;                       // zero shifted-overrun rows 736..751
    for (int i = tid; i < 128; i += 256) aw[(736 * 16) / 2 + i] = 0;

    // conv1 weights: k = tap'*2 + ch, tap' = ky*8 + kx (kx 0..7, 5 valid)
    for (int i = tid; i < 16 * 96; i += 256) {
      int oc = i / 96, k = i - oc * 96;
      int tp = k >> 1, c = k & 1;
      int ky = tp >> 3, kx = tp & 7;
      float v = 0.f;
      if (oc < 15 && kx < 5 && ky < 5)
        v = W1[oc * 75 + (c + 1) * 25 + ky * 5 + kx];
      w1p[i] = (_Float16)v;
    }
    // conv2 tap-pair weights: pair p -> taps 2p,2p+1; k<16: ic=k of tap0,
    // k>=16: ic=k-16 of tap1; ic=15 and tap 25 are zero padding.
    for (int i = tid; i < 13 * 16 * 32; i += 256) {
      int p = i >> 9, r = i & 511;
      int oc = r >> 5, k = r & 31;
      int t  = 2 * p + (k >> 4);
      int ic = k & 15;
      float v = 0.f;
      if (oc < 5 && ic < 15 && t < 25) v = W2[oc * 375 + ic * 25 + t];
      wb2[i] = (_Float16)v;
    }
    if (tid < 16) {  // fold time channel into conv1 bias
      float s = 0.f;
      if (tid < 15) {
        s = b1[tid];
        float ws = 0.f;
        for (int t = 0; t < 25; ++t) ws += W1[tid * 75 + t];
        s += tval * ws;
      }
      bias1_s[tid] = s;
    }
    for (int i = tid; i < 125; i += 256) w3s[i] = W3[i];
  }

  // ---- graph aggregation (register accumulate; no LDS writes yet)
  const int beg = off[node], end = off[node + 1];
  float acc0[4] = {0.f, 0.f, 0.f, 0.f}, acc1[4] = {0.f, 0.f, 0.f, 0.f};
  for (int i = beg; i < end; ++i) {
    int s = elist[i];
    float d0 = data[2 * s], d1 = data[2 * s + 1];
    const float* pr = posd + (size_t)s * HWP;
#pragma unroll
    for (int q = 0; q < 4; ++q) {
      int p = tid + q * 256;
      if (p < HWP) { float v = pr[p]; acc0[q] += d0 * v; acc1[q] += d1 * v; }
    }
  }
  __syncthreads();  // pads zeroed before valid cells written
  {
    v2h* xsp2w = (v2h*)xsp;
#pragma unroll
    for (int q = 0; q < 4; ++q) {
      int p = tid + q * 256;
      if (p < HWP) {
        int y = p / 20, x = p - y * 20;
        v2h v; v[0] = (_Float16)acc0[q]; v[1] = (_Float16)acc1[q];
        xsp2w[y * 24 + x] = v;            // single b32 store, channel pair
      }
    }
  }
  __syncthreads();

  // scalar wave id: all tile guards become SALU branches (EXEC untouched)
  const int w  = __builtin_amdgcn_readfirstlane(tid >> 5);
  const int lane = tid & 31, hl = lane >> 4, nl = lane & 15;
  const v8f z8 = {0.f, 0.f, 0.f, 0.f, 0.f, 0.f, 0.f, 0.f};
  const v2h* xsp2 = (const v2h*)xsp;

  // ---- conv1: M=736 (46 tiles, out width==16), N=16(15 oc), K=96 (3 ksteps)
  {
    v16h bf[3];
#pragma unroll
    for (int ks = 0; ks < 3; ++ks) {
      const _Float16* wb = &w1p[nl * 96 + ks * 32 + 8 * hl];
      bf[ks] = ld_frag(wb, wb + 16);
    }
    const float bias = bias1_s[nl];
    for (int tt = 0; tt < 6; ++tt) {
      const int ty = w + tt * 8;                    // scalar
      if (ty < 46) {                                // scalar branch
        // batch all 12 A-loads, then drain with 3 chained WMMAs
        const int xb = ty * 24 + nl + 4 * hl;
        v16h af0 = ld_a1frag(xsp2, xb);
        v16h af1 = ld_a1frag(xsp2, xb + 48);
        v16h af2 = ld_a1frag(xsp2, xb + 96);
        v8f c;
        c = __builtin_amdgcn_wmma_f32_16x16x32_f16(false, af0, false, bf[0],
                                                   (short)0, z8, false, false);
        c = __builtin_amdgcn_wmma_f32_16x16x32_f16(false, af1, false, bf[1],
                                                   (short)0, c, false, false);
        c = __builtin_amdgcn_wmma_f32_16x16x32_f16(false, af2, false, bf[2],
                                                   (short)0, c, false, false);
        // store position-major; oc=15 pad column is exact 0 only because
        // tanh(0+0)=0 (bias1_s[15]==0, B column was 0)
#pragma unroll
        for (int j = 0; j < 8; ++j) {
          int pos = ty * 16 + j + 8 * hl;
          a1p[pos * 16 + nl] = (_Float16)fast_tanh(c[j] + bias);
        }
      }
    }
  }
  __syncthreads();

  // ---- conv2: 13 tap-pair GEMMs, M=504 (32 tiles), N=16(5 oc), K=32/pair
  {
    int py[4], px[4];
#pragma unroll
    for (int tt = 0; tt < 4; ++tt) {
      int pos = (w + tt * 8) * 16 + nl;
      py[tt] = pos / 12;
      px[tt] = pos - py[tt] * 12;
    }
    v8f c[4] = {z8, z8, z8, z8};
#pragma unroll
    for (int p = 0; p < 13; ++p) {
      const int t0 = 2 * p,     ky0 = t0 / 5, kx0 = t0 - ky0 * 5;
      const int t1 = 2 * p + 1, ky1 = (t1 < 25) ? t1 / 5 : 5,
                                kx1 = (t1 < 25) ? t1 - (t1 / 5) * 5 : 0;
      // batch B + all 4 tile A-fragments, then 4 WMMAs
      const _Float16* wb = &wb2[(p * 16 + nl) * 32 + 8 * hl];
      v16h bf = ld_frag(wb, wb + 16);
      v16h afa[4];
#pragma unroll
      for (int tt = 0; tt < 4; ++tt) {
        int p0 = (py[tt] + ky0) * 16 + px[tt] + kx0;   // shifted conv1 positions
        int p1 = (py[tt] + ky1) * 16 + px[tt] + kx1;
        afa[tt] = ld_frag(&a1p[p0 * 16 + 8 * hl], &a1p[p1 * 16 + 8 * hl]);
      }
#pragma unroll
      for (int tt = 0; tt < 4; ++tt)
        c[tt] = __builtin_amdgcn_wmma_f32_16x16x32_f16(
            false, afa[tt], false, bf, (short)0, c[tt], false, false);
    }
    float bias = b2[(nl < 5) ? nl : 0];
    for (int tt = 0; tt < 4; ++tt) {
#pragma unroll
      for (int j = 0; j < 8; ++j) {
        int pos = (w + tt * 8) * 16 + j + 8 * hl;
        if (nl < 5 && pos < C2P) a2[nl * C2P + pos] = c[tt][j] + bias;
      }
    }
  }
  __syncthreads();

  // ---- conv3: 1 out channel, K=125 (small -> VALU, all-constant indexing)
  for (int p = tid; p < C3P; p += 256) {
    int y = p >> 3, x = p & 7;
    float s = b3[0];
#pragma unroll
    for (int ic = 0; ic < 5; ++ic)
#pragma unroll
      for (int ky = 0; ky < 5; ++ky)
#pragma unroll
        for (int kx = 0; kx < 5; ++kx)
          s += a2[ic * C2P + (y + ky) * 12 + (x + kx)] * w3s[ic * 25 + ky * 5 + kx];
    a3[p] = s;
  }
  __syncthreads();

  // ---- final linear 304 -> 2
  if (tid < 2) {
    float s = bl[tid];
    for (int i = 0; i < C3P; ++i) s += a3[i] * Wl[tid * C3P + i];
    out[node * 2 + tid] = s;
  }
}

// ---------------- host launch ----------------

extern "C" void kernel_launch(void* const* d_in, const int* in_sizes, int n_in,
                              void* d_out, int out_size, void* d_ws, size_t ws_size,
                              hipStream_t stream) {
  (void)n_in; (void)out_size; (void)ws_size;
  const float*     tptr  = (const float*)d_in[0];
  const float*     data  = (const float*)d_in[1];
  const long long* edges = (const long long*)d_in[2];   // int64 [2,E]
  const float*     posd  = (const float*)d_in[3];
  const float* W1 = (const float*)d_in[4];
  const float* b1 = (const float*)d_in[5];
  const float* W2 = (const float*)d_in[6];
  const float* b2 = (const float*)d_in[7];
  const float* W3 = (const float*)d_in[8];
  const float* b3 = (const float*)d_in[9];
  const float* Wl = (const float*)d_in[10];
  const float* bl = (const float*)d_in[11];
  float* out = (float*)d_out;

  const int n = in_sizes[3] / HWP;   // 10000
  const int e = in_sizes[2] / 2;     // 90000

  char*  p   = (char*)d_ws;
  size_t szA = (((size_t)n * 4) + 255) & ~(size_t)255;
  size_t szB = (((size_t)(n + 1) * 4) + 255) & ~(size_t)255;
  int* cnt   = (int*)p; p += szA;
  int* off   = (int*)p; p += szB;
  int* cur   = (int*)p; p += szA;
  int* elist = (int*)p;              // e + n entries

  k_init_cnt<<<(n + 255) / 256, 256, 0, stream>>>(cnt, n);
  k_count<<<(e + 255) / 256, 256, 0, stream>>>(edges, cnt, e);
  k_scan<<<1, 1024, 0, stream>>>(cnt, off, cur, n);
  k_fill<<<(e + n + 255) / 256, 256, 0, stream>>>(edges, cur, elist, e, n);
  k_fused<<<n, 256, 0, stream>>>(tptr, data, posd, W1, b1, W2, b2, W3, b3,
                                 Wl, bl, off, elist, out);
}
